// DSRAModel_20074677142347
// MI455X (gfx1250) — matrix operand order repaired and checked by reading, not compiled, not run on hardware
//
#include <hip/hip_runtime.h>
#include <hip/hip_bf16.h>
#include <math.h>

typedef unsigned short ushort_t;
typedef __attribute__((ext_vector_type(16))) __bf16 v16bf;
typedef __attribute__((ext_vector_type(8)))  float  v8f;
typedef __attribute__((ext_vector_type(4)))  int    v4i;

#define BB   2
#define SS   4096
#define DDIM 1024
#define KDIM 128
#define VOC  32000
#define NH   8
#define CHK  256
#define NCH  16
#define DKH  16
#define DVH  128
#define MM   (BB*SS)

// CDNA5 async global->LDS path (ASYNCcnt), guarded so absence degrades gracefully.
#if defined(__gfx1250__) && \
    __has_builtin(__builtin_amdgcn_global_load_async_to_lds_b128) && \
    __has_builtin(__builtin_amdgcn_s_wait_asynccnt)
#define USE_ASYNC_LDS 1
#define ASYNC_CP_B128(g, l, off) \
    __builtin_amdgcn_global_load_async_to_lds_b128( \
        (__attribute__((address_space(1))) v4i*)(g), \
        (__attribute__((address_space(3))) v4i*)(l), (off), 0)
#else
#define USE_ASYNC_LDS 0
#endif

__device__ __forceinline__ ushort_t f2bf(float f) {
    unsigned int u = __float_as_uint(f);
    unsigned int r = u + 0x7fffu + ((u >> 16) & 1u);   // round-to-nearest-even
    return (ushort_t)(r >> 16);
}

// ---------------- f32 [K,N] -> bf16 [N,K] tiled transpose-convert (weights)
__global__ __launch_bounds__(256)
void transpose_convert(const float* __restrict__ in, ushort_t* __restrict__ out,
                       int K, int N) {
    __shared__ float tile[32][33];
    int kb = blockIdx.y * 32, nb = blockIdx.x * 32;
    int tx = threadIdx.x & 31, ty = threadIdx.x >> 5;        // 32 x 8
#pragma unroll
    for (int r = ty; r < 32; r += 8)
        tile[r][tx] = in[(size_t)(kb + r) * N + nb + tx];
    __syncthreads();
#pragma unroll
    for (int r = ty; r < 32; r += 8)
        out[(size_t)(nb + r) * K + kb + tx] = f2bf(tile[tx][r]);
}

// ------------------------------------------------- embedding + 4-tap context
__global__ __launch_bounds__(256)
void embed_ctx_kernel(const int* __restrict__ x, const float* __restrict__ emb,
                      float* __restrict__ ctx, ushort_t* __restrict__ ctxb) {
    int row = blockIdx.x;                 // 0..MM-1
    int s = row % SS;
    int t0 = x[row];
    int t1 = (s >= 1) ? x[row - 1] : -1;
    int t2 = (s >= 2) ? x[row - 2] : -1;
    int t3 = (s >= 3) ? x[row - 3] : -1;
    for (int d = threadIdx.x; d < DDIM; d += 256) {
        float v = emb[(size_t)t0 * DDIM + d];
        if (t1 >= 0) v += emb[(size_t)t1 * DDIM + d];
        if (t2 >= 0) v += emb[(size_t)t2 * DDIM + d];
        if (t3 >= 0) v += emb[(size_t)t3 * DDIM + d];
        ctx[(size_t)row * DDIM + d]  = v;
        ctxb[(size_t)row * DDIM + d] = f2bf(v);
    }
}

// ---------------------------------------------------------------- WMMA GEMM
#define BM 128
#define BN 128
#define BKT 64

constexpr int EPI_NONE = 0;
constexpr int EPI_PHI  = 1;   // elu(x)+1
constexpr int EPI_RES  = 2;   // + residual[M,N]
constexpr int EPI_BIAS = 3;   // + bias[N], nontemporal store

union FragAB { v16bf v; uint4 q[2]; ushort_t s[16]; };

// A: bf16 [M,K] row-major.  Bt: bf16 [N,K] (pre-transposed weights).
template<int EPI>
__global__ __launch_bounds__(256)
void gemm_bf16(const ushort_t* __restrict__ A, const ushort_t* __restrict__ Bt,
               float* __restrict__ C, const float* __restrict__ extra,
               int M, int N, int Kdim) {
    __shared__ ushort_t sA[2][BM][BKT];   // 2 x 16 KB
    __shared__ ushort_t sB[2][BN][BKT];   // 2 x 16 KB

    int tid  = threadIdx.x;
    int lane = tid & 31;
    int wave = tid >> 5;
    int wm = wave & 3;                    // 4 waves down M
    int wn = wave >> 2;                   // 2 waves across N
    int rowblk = blockIdx.y * BM;
    int colblk = blockIdx.x * BN;

    v8f acc[2][4];
    v8f zero = {};
#pragma unroll
    for (int i = 0; i < 2; i++)
#pragma unroll
        for (int j = 0; j < 4; j++) acc[i][j] = zero;

    // staging map: each thread copies 32 contiguous bf16 of one row (A and B)
    int srow = tid >> 1;                  // 0..127
    int scol = (tid & 1) * 32;            // 0 / 32
    const ushort_t* gA = A  + (size_t)(rowblk + srow) * Kdim + scol;
    const ushort_t* gB = Bt + (size_t)(colblk + srow) * Kdim + scol;

    auto stage = [&](int buf, int k0) {
#if USE_ASYNC_LDS
        ASYNC_CP_B128(gA + k0, &sA[buf][srow][scol], 0);
        ASYNC_CP_B128(gA + k0, &sA[buf][srow][scol], 16);
        ASYNC_CP_B128(gA + k0, &sA[buf][srow][scol], 32);
        ASYNC_CP_B128(gA + k0, &sA[buf][srow][scol], 48);
        ASYNC_CP_B128(gB + k0, &sB[buf][srow][scol], 0);
        ASYNC_CP_B128(gB + k0, &sB[buf][srow][scol], 16);
        ASYNC_CP_B128(gB + k0, &sB[buf][srow][scol], 32);
        ASYNC_CP_B128(gB + k0, &sB[buf][srow][scol], 48);
#else
        const uint4* a4 = reinterpret_cast<const uint4*>(gA + k0);
        uint4* da = reinterpret_cast<uint4*>(&sA[buf][srow][scol]);
        da[0] = a4[0]; da[1] = a4[1]; da[2] = a4[2]; da[3] = a4[3];
        const uint4* b4 = reinterpret_cast<const uint4*>(gB + k0);
        uint4* db = reinterpret_cast<uint4*>(&sB[buf][srow][scol]);
        db[0] = b4[0]; db[1] = b4[1]; db[2] = b4[2]; db[3] = b4[3];
#endif
    };

    int nk = Kdim / BKT;
    stage(0, 0);
#if USE_ASYNC_LDS
    __builtin_amdgcn_s_wait_asynccnt(0);
#endif
    __syncthreads();

    int m_    = lane & 15;
    int khalf = (lane >> 4) * 8;          // A k-half select (ISA 7.12.2)
    int kb    = (lane >> 4) * 8;          // B k-group select

    for (int kt = 0; kt < nk; kt++) {
        int cb = kt & 1;
        if (kt + 1 < nk) stage(cb ^ 1, (kt + 1) * BKT);   // prefetch next tile
#pragma unroll
        for (int ks = 0; ks < 2; ks++) {                  // two WMMA K-steps
            FragAB af[2];
#pragma unroll
            for (int i = 0; i < 2; i++) {
                int r = wm * 32 + i * 16 + m_;
                af[i].q[0] = *reinterpret_cast<const uint4*>(&sA[cb][r][ks * 32 + khalf]);
                af[i].q[1] = *reinterpret_cast<const uint4*>(&sA[cb][r][ks * 32 + 16 + khalf]);
            }
#pragma unroll
            for (int j = 0; j < 4; j++) {
                FragAB bfr;
                int n_ = wn * 64 + j * 16 + m_;
                bfr.q[0] = *reinterpret_cast<const uint4*>(&sB[cb][n_][ks * 32 + kb]);
                bfr.q[1] = *reinterpret_cast<const uint4*>(&sB[cb][n_][ks * 32 + 16 + kb]);
#pragma unroll
                for (int i = 0; i < 2; i++)
                    acc[i][j] = __builtin_amdgcn_wmma_f32_16x16x32_bf16(
                        false, af[i].v, false, bfr.v, (short)0, acc[i][j], false, false);
            }
        }
#if USE_ASYNC_LDS
        __builtin_amdgcn_s_wait_asynccnt(0);
#endif
        __syncthreads();
    }

    // store D (C/D layout: m = r + 8*(lane/16), n = lane%16)
    int mh = (lane >> 4) * 8;
#pragma unroll
    for (int i = 0; i < 2; i++) {
#pragma unroll
        for (int j = 0; j < 4; j++) {
            int gcol = colblk + wn * 64 + j * 16 + m_;
#pragma unroll
            for (int r = 0; r < 8; r++) {
                int grow = rowblk + wm * 32 + i * 16 + mh + r;
                float val = acc[i][j][r];
                if (EPI == EPI_PHI)       val = (val > 0.f) ? val + 1.f : __expf(val);
                else if (EPI == EPI_RES)  val += extra[(size_t)grow * N + gcol];
                else if (EPI == EPI_BIAS) val += extra[gcol];
                if (EPI == EPI_BIAS)
                    __builtin_nontemporal_store(val, &C[(size_t)grow * N + gcol]);
                else
                    C[(size_t)grow * N + gcol] = val;
            }
        }
    }
}

// ------------------------------------------------------- chunked linear-attn
__global__ __launch_bounds__(256)
void scan_kernel(const float* __restrict__ q, const float* __restrict__ k,
                 const float* __restrict__ v, ushort_t* __restrict__ opre) {
    __shared__ float sk[2][CHK][DKH];   // 32 KB : current/prev chunk keys
    __shared__ float sS[DKH][DVH];      //  8 KB : outer-product state
    __shared__ float sT[DKH];           //        running key sum
    int b = blockIdx.x >> 3;
    int h = blockIdx.x & 7;
    int c = threadIdx.x;                // row within chunk

    for (int t = threadIdx.x; t < DKH * DVH; t += 256) ((float*)sS)[t] = 0.f;
    for (int t = threadIdx.x; t < CHK * DKH; t += 256) ((float*)sk[1])[t] = 0.f;
    if (threadIdx.x < DKH) sT[threadIdx.x] = 0.f;
    __syncthreads();

    for (int i = 0; i < NCH; i++) {
        int cur = i & 1, prv = cur ^ 1;
        {   // load this chunk's keys (thread c loads row c)
            const float4* krow = reinterpret_cast<const float4*>(
                k + (size_t)(b * SS + i * CHK + c) * KDIM + h * DKH);
            float4* dst = reinterpret_cast<float4*>(&sk[cur][c][0]);
#pragma unroll
            for (int kk = 0; kk < 4; kk++) dst[kk] = krow[kk];
        }
        float qr[DKH];
        {
            const float* qrow = q + (size_t)(b * SS + i * CHK + c) * KDIM + h * DKH;
#pragma unroll
            for (int kk = 0; kk < DKH; kk++) qr[kk] = qrow[kk];
        }
        __syncthreads();

        float den = 0.f;
#pragma unroll
        for (int kk = 0; kk < DKH; kk++) den += qr[kk] * sT[kk];

        const float* vbase = v + (size_t)(b * SS + i * CHK) * DDIM + h * DVH;
        size_t orow = (size_t)(b * SS + i * CHK + c) * DDIM + h * DVH;

        float acc[64];
        // ---- pass 0: v columns 0..63, also finish the denominator
#pragma unroll
        for (int j = 0; j < 64; j++) {
            float sum = 0.f;
#pragma unroll
            for (int kk = 0; kk < DKH; kk++) sum += qr[kk] * sS[kk][j];
            acc[j] = sum;
        }
        for (int s = 0; s < CHK; s++) {
            float att = 0.f, byp = 0.f;
#pragma unroll
            for (int kk = 0; kk < DKH; kk++) {
                att += qr[kk] * sk[cur][s][kk];
                byp += qr[kk] * sk[prv][s][kk];
            }
            if (s > c) att = 0.f;               // causal mask
            float w = att + byp;
            den += w;                           // = q.kcum + q.(sum bk)
            const float4* vr = reinterpret_cast<const float4*>(vbase + (size_t)s * DDIM);
#pragma unroll
            for (int j4 = 0; j4 < 16; j4++) {
                float4 vv = vr[j4];
                acc[4*j4+0] += w * vv.x; acc[4*j4+1] += w * vv.y;
                acc[4*j4+2] += w * vv.z; acc[4*j4+3] += w * vv.w;
            }
        }
        float inv = 1.f / (den + 1e-6f);
        {
            unsigned* op = reinterpret_cast<unsigned*>(opre + orow);
#pragma unroll
            for (int j = 0; j < 32; j++)
                op[j] = (unsigned)f2bf(acc[2*j] * inv) |
                        ((unsigned)f2bf(acc[2*j+1] * inv) << 16);
        }

        // ---- pass 1: v columns 64..127 (den already final)
#pragma unroll
        for (int j = 0; j < 64; j++) {
            float sum = 0.f;
#pragma unroll
            for (int kk = 0; kk < DKH; kk++) sum += qr[kk] * sS[kk][64 + j];
            acc[j] = sum;
        }
        for (int s = 0; s < CHK; s++) {
            float att = 0.f, byp = 0.f;
#pragma unroll
            for (int kk = 0; kk < DKH; kk++) {
                att += qr[kk] * sk[cur][s][kk];
                byp += qr[kk] * sk[prv][s][kk];
            }
            if (s > c) att = 0.f;
            float w = att + byp;
            const float4* vr = reinterpret_cast<const float4*>(vbase + (size_t)s * DDIM + 64);
#pragma unroll
            for (int j4 = 0; j4 < 16; j4++) {
                float4 vv = vr[j4];
                acc[4*j4+0] += w * vv.x; acc[4*j4+1] += w * vv.y;
                acc[4*j4+2] += w * vv.z; acc[4*j4+3] += w * vv.w;
            }
        }
        {
            unsigned* op = reinterpret_cast<unsigned*>(opre + orow + 64);
#pragma unroll
            for (int j = 0; j < 32; j++)
                op[j] = (unsigned)f2bf(acc[2*j] * inv) |
                        ((unsigned)f2bf(acc[2*j+1] * inv) << 16);
        }

        __syncthreads();
        // fold previous chunk into the long-range state
        if (i >= 1) {
            int kk  = threadIdx.x & 15;
            int vb0 = (threadIdx.x >> 4) * 8;
            const float* vprev =
                v + (size_t)(b * SS + (i - 1) * CHK) * DDIM + h * DVH + vb0;
            float a[8];
#pragma unroll
            for (int j = 0; j < 8; j++) a[j] = 0.f;
            float st = 0.f;
            for (int s = 0; s < CHK; s++) {
                float kv = sk[prv][s][kk];
                st += kv;
                const float4* vr = reinterpret_cast<const float4*>(vprev + (size_t)s * DDIM);
                float4 v0 = vr[0], v1 = vr[1];
                a[0] += kv * v0.x; a[1] += kv * v0.y; a[2] += kv * v0.z; a[3] += kv * v0.w;
                a[4] += kv * v1.x; a[5] += kv * v1.y; a[6] += kv * v1.z; a[7] += kv * v1.w;
            }
#pragma unroll
            for (int j = 0; j < 8; j++) sS[kk][vb0 + j] += a[j];
            if (vb0 == 0) sT[kk] += st;
        }
        __syncthreads();
    }
}

// ----------------------------------------------------------------- LayerNorm
__global__ __launch_bounds__(256)
void layernorm_kernel(const float* __restrict__ y, const float* __restrict__ g,
                      const float* __restrict__ bta, ushort_t* __restrict__ h) {
    __shared__ float red[256];
    int row = blockIdx.x;
    const float* yr = y + (size_t)row * DDIM;
    float s = 0.f, s2 = 0.f, vals[4];
#pragma unroll
    for (int j = 0; j < 4; j++) {
        float v_ = yr[threadIdx.x + j * 256];
        vals[j] = v_;
        s += v_; s2 += v_ * v_;
    }
    red[threadIdx.x] = s; __syncthreads();
    for (int off = 128; off > 0; off >>= 1) {
        if (threadIdx.x < off) red[threadIdx.x] += red[threadIdx.x + off];
        __syncthreads();
    }
    float mean = red[0] / DDIM; __syncthreads();
    red[threadIdx.x] = s2; __syncthreads();
    for (int off = 128; off > 0; off >>= 1) {
        if (threadIdx.x < off) red[threadIdx.x] += red[threadIdx.x + off];
        __syncthreads();
    }
    float var  = red[0] / DDIM - mean * mean;
    float rstd = rsqrtf(var + 1e-5f);
#pragma unroll
    for (int j = 0; j < 4; j++) {
        int d = threadIdx.x + j * 256;
        h[(size_t)row * DDIM + d] = f2bf((vals[j] - mean) * rstd * g[d] + bta[d]);
    }
}

// ------------------------------------------------------------------- launch
extern "C" void kernel_launch(void* const* d_in, const int* in_sizes, int n_in,
                              void* d_out, int out_size, void* d_ws, size_t ws_size,
                              hipStream_t stream) {
    const int*   x    = (const int*)d_in[0];
    const float* emb  = (const float*)d_in[1];
    const float* Wq   = (const float*)d_in[2];
    const float* Wk   = (const float*)d_in[3];
    const float* Wv   = (const float*)d_in[4];
    const float* Wo   = (const float*)d_in[5];
    const float* ln_g = (const float*)d_in[6];
    const float* ln_b = (const float*)d_in[7];
    const float* outW = (const float*)d_in[8];
    const float* outb = (const float*)d_in[9];

    char* ws = (char*)d_ws;
    size_t off = 0;
    auto alloc = [&](size_t bytes) -> char* {
        char* p = ws + off;
        off += (bytes + 255) & ~(size_t)255;
        return p;
    };
    float*    ctx  = (float*)   alloc((size_t)MM * DDIM * 4);
    ushort_t* ctxb = (ushort_t*)alloc((size_t)MM * DDIM * 2);
    float*    qf   = (float*)   alloc((size_t)MM * KDIM * 4);
    float*    kf   = (float*)   alloc((size_t)MM * KDIM * 4);
    float*    vf   = (float*)   alloc((size_t)MM * DDIM * 4);
    ushort_t* opre = (ushort_t*)alloc((size_t)MM * DDIM * 2);
    float*    yf   = (float*)   alloc((size_t)MM * DDIM * 4);
    ushort_t* hb   = (ushort_t*)alloc((size_t)MM * DDIM * 2);
    ushort_t* wqb  = (ushort_t*)alloc((size_t)DDIM * KDIM * 2);   // [N=128][K=1024]
    ushort_t* wkb  = (ushort_t*)alloc((size_t)DDIM * KDIM * 2);
    ushort_t* wvb  = (ushort_t*)alloc((size_t)DDIM * DDIM * 2);
    ushort_t* wob  = (ushort_t*)alloc((size_t)DDIM * DDIM * 2);
    ushort_t* owb  = (ushort_t*)alloc((size_t)DDIM * VOC  * 2);   // [N=32000][K=1024]

    // weights: f32 [K,N] -> bf16 [N,K]
    transpose_convert<<<dim3(KDIM / 32, DDIM / 32), 256, 0, stream>>>(Wq,  wqb, DDIM, KDIM);
    transpose_convert<<<dim3(KDIM / 32, DDIM / 32), 256, 0, stream>>>(Wk,  wkb, DDIM, KDIM);
    transpose_convert<<<dim3(DDIM / 32, DDIM / 32), 256, 0, stream>>>(Wv,  wvb, DDIM, DDIM);
    transpose_convert<<<dim3(DDIM / 32, DDIM / 32), 256, 0, stream>>>(Wo,  wob, DDIM, DDIM);
    transpose_convert<<<dim3(VOC  / 32, DDIM / 32), 256, 0, stream>>>(outW, owb, DDIM, VOC);

    embed_ctx_kernel<<<MM, 256, 0, stream>>>(x, emb, ctx, ctxb);

    gemm_bf16<EPI_PHI><<<dim3(KDIM / BN, MM / BM), 256, 0, stream>>>(
        ctxb, wqb, qf, nullptr, MM, KDIM, DDIM);
    gemm_bf16<EPI_PHI><<<dim3(KDIM / BN, MM / BM), 256, 0, stream>>>(
        ctxb, wkb, kf, nullptr, MM, KDIM, DDIM);
    gemm_bf16<EPI_NONE><<<dim3(DDIM / BN, MM / BM), 256, 0, stream>>>(
        ctxb, wvb, vf, nullptr, MM, DDIM, DDIM);

    scan_kernel<<<BB * NH, 256, 0, stream>>>(qf, kf, vf, opre);

    gemm_bf16<EPI_RES><<<dim3(DDIM / BN, MM / BM), 256, 0, stream>>>(
        opre, wob, yf, ctx, MM, DDIM, DDIM);

    layernorm_kernel<<<MM, 256, 0, stream>>>(yf, ln_g, ln_b, hb);

    gemm_bf16<EPI_BIAS><<<dim3(VOC / BN, MM / BM), 256, 0, stream>>>(
        hb, owb, (float*)d_out, outb, MM, VOC, DDIM);
}
